// SelfAttention_34119220199937
// MI455X (gfx1250) — matrix operand order repaired and checked by reading, not compiled, not run on hardware
//
#include <hip/hip_runtime.h>
#include <hip/hip_bf16.h>

typedef __attribute__((ext_vector_type(16))) _Float16 v16h;
typedef __attribute__((ext_vector_type(8)))  _Float16 v8h;
typedef __attribute__((ext_vector_type(8)))  float    v8f;
typedef __attribute__((ext_vector_type(4)))  unsigned int u32x4;
typedef __attribute__((ext_vector_type(8)))  int i32x8;
typedef __attribute__((ext_vector_type(4)))  int i32x4;

#define WMMA_F16(a, b, c) \
  __builtin_amdgcn_wmma_f32_16x16x32_f16(false, (a), false, (b), (short)0, (c), false, false)

// ---------- dimensions: B=2, S=2048, D=1024, H=16, hd=64, M=4096 ----------
#define DIM_S 2048
#define DIM_D 1024
#define DIM_H 16
#define DIM_HD 64
#define DIM_M 4096

// Decisive TDM probe: device pass MUST have the TDM builtins, else fail loudly.
#if defined(__gfx1250__)
#if !(__has_builtin(__builtin_amdgcn_tensor_load_to_lds) && __has_builtin(__builtin_amdgcn_s_wait_tensorcnt))
#error "CDNA5 probe: __builtin_amdgcn_tensor_load_to_lds / s_wait_tensorcnt NOT available on this toolchain"
#endif
#endif

// ---------- fragment loaders (layouts per CDNA5 ISA 7.12.2) ----------
__device__ __forceinline__ v16h load_a_frag(const _Float16* base, int ld) {
  const int lane = threadIdx.x & 31;
  const _Float16* p = base + (lane & 15) * ld + ((lane < 16) ? 0 : 8);
  union { v16h v; v8h h[2]; } u;
  u.h[0] = *(const v8h*)(p);
  u.h[1] = *(const v8h*)(p + 16);
  return u.v;
}
__device__ __forceinline__ v16h load_b_frag(const _Float16* base, int ld) {
  const int lane = threadIdx.x & 31;
  const _Float16* p = base + (lane & 15) * ld + ((lane < 16) ? 0 : 8);
  union { v16h v; v8h h[2]; } u;
  u.h[0] = *(const v8h*)(p);
  u.h[1] = *(const v8h*)(p + 16);
  return u.v;
}

__device__ __forceinline__ float smax_s(float x) {
  if (x >= 0.0f) return 1.0f + x * (1.0f + 0.5f * x * (1.0f + x * (1.0f / 3.0f)));
  return 1.0f / (1.0f - x * (1.0f - 0.5f * x * (1.0f - x * (1.0f / 3.0f))));
}

// ---------- TDM staging: ROWS x 1024 f16 panel, global -> LDS ----------
template<int ROWS>
__device__ __forceinline__ void stage_a_panel(_Float16* shA, const _Float16* Ar, int tid) {
#if defined(__gfx1250__)
  if (tid == 0) {   // one wave issues; TDM ignores EXEC
    const unsigned long long ga = (unsigned long long)(size_t)Ar;
    const unsigned lds = (unsigned)(size_t)shA;   // LDS aperture: low 32 bits = LDS byte addr
    u32x4 g0 = {};
    g0[0] = 1u;                                   // count=1 (valid), user mode
    g0[1] = lds;                                  // lds_addr
    g0[2] = (unsigned)(ga & 0xffffffffu);         // global_addr[31:0]
    g0[3] = (unsigned)((ga >> 32) & 0x1ffffffu) | (2u << 30);  // addr[56:32] | type=2
    i32x8 g1 = {};
    g1[0] = (1 << 16);                            // workgroup_mask=0, data_size=1 (2 bytes)
    g1[1] = (int)(1024u << 16);                   // tensor_dim0 = 1024
    g1[2] = (int)(4096u << 16);                   // tensor_dim1 = 4096
    g1[3] = (int)(1024u << 16);                   // tile_dim0 = 1024
    g1[4] = ROWS;                                 // tile_dim1 = ROWS
    g1[5] = 1024;                                 // tensor_dim0_stride
    i32x4 z4 = {};
#if __clang_major__ >= 23
    i32x8 z8 = {};
    __builtin_amdgcn_tensor_load_to_lds(g0, g1, z4, z4, z8, 0);
#else
    __builtin_amdgcn_tensor_load_to_lds(g0, g1, z4, z4, 0);
#endif
    __builtin_amdgcn_s_wait_tensorcnt(0);
  }
#else
  for (int i = tid; i < ROWS * DIM_D / 8; i += 256)
    ((v8h*)shA)[i] = ((const v8h*)Ar)[i];
#endif
  __syncthreads();
}

// ---------- GEMM body: 64M x 64N per wave (4 m-tiles x 4 n-tiles), double-buffered B ----------
__device__ __forceinline__ void gemm64x512_panel(const _Float16* shA, const _Float16* W,
                                                 size_t n0, v8f (&acc)[4][4]) {
  v16h bbuf[2][4];
#pragma unroll
  for (int j = 0; j < 4; ++j)
    bbuf[0][j] = load_b_frag(W + (n0 + j * 16) * DIM_D, DIM_D);
#pragma unroll 2
  for (int k = 0; k < DIM_D; k += 32) {
    const int cur = (k >> 5) & 1, nxt = cur ^ 1;
    if (k + 32 < DIM_D) {
#pragma unroll
      for (int j = 0; j < 4; ++j)
        bbuf[nxt][j] = load_b_frag(W + (n0 + j * 16) * DIM_D + k + 32, DIM_D);
    }
#pragma unroll
    for (int mt = 0; mt < 4; ++mt) {
      v16h a = load_a_frag(shA + mt * 16 * DIM_D + k, DIM_D);
#pragma unroll
      for (int j = 0; j < 4; ++j)
        acc[mt][j] = WMMA_F16(a, bbuf[cur][j], acc[mt][j]);
    }
  }
}

#define GEMM_SMEM (64 * DIM_D * 2)   // 128 KB A panel

__global__ void f32_to_f16_kernel(const float* __restrict__ in, _Float16* __restrict__ out, int n) {
  int i = blockIdx.x * blockDim.x + threadIdx.x;
  if (i < n) out[i] = (_Float16)in[i];
}

// C[4096x3072] = Xh * Wqkv^T ; scatter Q[b,h,s,e]*scale, K[b,h,s,e], Vt[b,h,e,s]
__global__ __launch_bounds__(256) void gemm_qkv_kernel(
    const _Float16* __restrict__ X, const _Float16* __restrict__ W,
    _Float16* __restrict__ Qf, _Float16* __restrict__ Kf, _Float16* __restrict__ Vt) {
  extern __shared__ char gsm[];
  _Float16* shA = (_Float16*)gsm;
  const int tid = threadIdx.x;
  const int w = tid >> 5, lane = tid & 31;
  const int mbase = (lane < 16) ? 0 : 8;
  const int nlane = lane & 15;
  const int m0 = blockIdx.y * 64;
  const size_t n0 = blockIdx.x * 512 + w * 64;

  stage_a_panel<64>(shA, X + (size_t)m0 * DIM_D, tid);

  v8f acc[4][4] = {};
  gemm64x512_panel(shA, W, n0, acc);

#pragma unroll
  for (int mt = 0; mt < 4; ++mt) {
#pragma unroll
    for (int j = 0; j < 4; ++j) {
      const int n = (int)n0 + j * 16 + nlane;
      const int c = n >> 10;          // 0=q,1=k,2=v
      const int d = n & 1023;
      const int h = d >> 6, e = d & 63;
#pragma unroll
      for (int r = 0; r < 8; ++r) {
        const int m = m0 + mt * 16 + mbase + r;
        const int bb = m >> 11, s = m & 2047;
        const float v = acc[mt][j][r];
        const size_t bh = (size_t)(bb * DIM_H + h);
        if (c == 0)      Qf[(bh * DIM_S + s) * DIM_HD + e] = (_Float16)(v * 0.125f);
        else if (c == 1) Kf[(bh * DIM_S + s) * DIM_HD + e] = (_Float16)v;
        else             Vt[(bh * DIM_HD + e) * DIM_S + s] = (_Float16)v;
      }
    }
  }
}

// ---------- attention: one workgroup = (b,h) x 16-query tile ----------
#define SMEM_SC    0
#define SMEM_PR    (16 * DIM_S * 4)                 // 131072
#define SMEM_OACC  (SMEM_PR + 16 * DIM_S * 2)       // +65536
#define SMEM_RSUM  (SMEM_OACC + 16 * DIM_HD * 4)    // +4096
#define SMEM_TOTAL (SMEM_RSUM + 16 * 4)             // 200768 bytes

__global__ __launch_bounds__(256) void attn_kernel(
    const _Float16* __restrict__ Qf, const _Float16* __restrict__ Kf,
    const _Float16* __restrict__ Vt, _Float16* __restrict__ Oh) {
  extern __shared__ char smem[];
  float*    sc     = (float*)(smem + SMEM_SC);
  _Float16* pr     = (_Float16*)(smem + SMEM_PR);
  float*    oacc   = (float*)(smem + SMEM_OACC);
  float*    rowsum = (float*)(smem + SMEM_RSUM);

  const int qt = blockIdx.x, bh = blockIdx.y;
  const int s0 = qt * 16;
  const int tid = threadIdx.x;
  const int w = tid >> 5, lane = tid & 31;
  const int mbase = (lane < 16) ? 0 : 8;
  const int nlane = lane & 15;

  const _Float16* Qb = Qf + ((size_t)bh * DIM_S + s0) * DIM_HD;
  const _Float16* Kb = Kf + (size_t)bh * DIM_S * DIM_HD;
  const _Float16* Vb = Vt + (size_t)bh * DIM_HD * DIM_S;

  for (int i = tid; i < 16 * DIM_HD; i += 256) oacc[i] = 0.0f;

  // ---- Phase 1: scores = Q(16x64) * K^T(64x2048); double-buffered K fragments
  v16h a0 = load_a_frag(Qb + 0, DIM_HD);
  v16h a1 = load_a_frag(Qb + 32, DIM_HD);
  v16h b0 = load_b_frag(Kb + (size_t)(w * 16) * DIM_HD + 0, DIM_HD);
  v16h b1 = load_b_frag(Kb + (size_t)(w * 16) * DIM_HD + 32, DIM_HD);
  for (int nt = w; nt < DIM_S / 16; nt += 8) {
    v16h nb0 = b0, nb1 = b1;
    if (nt + 8 < DIM_S / 16) {
      const size_t tn = (size_t)(nt + 8) * 16 * DIM_HD;
      nb0 = load_b_frag(Kb + tn + 0, DIM_HD);
      nb1 = load_b_frag(Kb + tn + 32, DIM_HD);
    }
    v8f acc = {};
    acc = WMMA_F16(a0, b0, acc);
    acc = WMMA_F16(a1, b1, acc);
    const int t0 = nt * 16;
#pragma unroll
    for (int r = 0; r < 8; ++r)
      sc[(mbase + r) * DIM_S + t0 + nlane] = acc[r];
    b0 = nb0; b1 = nb1;
  }
  __syncthreads();

  // ---- Phase 2: per-row max, s(x-max), row sum (normalization deferred), probs -> f16
#pragma unroll
  for (int rr = 0; rr < 2; ++rr) {
    const int row = w * 2 + rr;
    float mx = -3.4e38f;
    for (int t = lane; t < DIM_S; t += 32) mx = fmaxf(mx, sc[row * DIM_S + t]);
#pragma unroll
    for (int off = 16; off > 0; off >>= 1) mx = fmaxf(mx, __shfl_xor(mx, off, 32));
    float sum = 0.0f;
    for (int t = lane; t < DIM_S; t += 32) {
      float s = smax_s(sc[row * DIM_S + t] - mx);
      sum += s;
      pr[row * DIM_S + t] = (_Float16)s;
    }
#pragma unroll
    for (int off = 16; off > 0; off >>= 1) sum += __shfl_xor(sum, off, 32);
    if (lane == 0) rowsum[row] = sum;
  }
  __syncthreads();

  // ---- Phase 3: O(16x64) = P(16x2048) * V; contraction split over 8 waves
  v8f pacc[4] = {};
  const int kbeg = w * 256;
  for (int kc = kbeg; kc < kbeg + 256; kc += 32) {
    v16h b[4];
#pragma unroll
    for (int j = 0; j < 4; ++j)
      b[j] = load_b_frag(Vb + (size_t)(j * 16) * DIM_S + kc, DIM_S);
    v16h a = load_a_frag(pr + kc, DIM_S);
#pragma unroll
    for (int j = 0; j < 4; ++j)
      pacc[j] = WMMA_F16(a, b[j], pacc[j]);
  }
#pragma unroll
  for (int j = 0; j < 4; ++j)
#pragma unroll
    for (int r = 0; r < 8; ++r)
      atomicAdd(&oacc[(mbase + r) * DIM_HD + j * 16 + nlane], pacc[j][r]);
  __syncthreads();

  // ---- epilogue: normalize, write attention out [B,S,D] f16
  const int b = bh >> 4, h = bh & 15;
  for (int i = tid; i < 16 * DIM_HD; i += 256) {
    const int m = i >> 6, e = i & 63;
    const float val = oacc[i] / rowsum[m];
    Oh[((size_t)(b * DIM_S + s0 + m)) * DIM_D + h * DIM_HD + e] = (_Float16)val;
  }
}

// out[4096x1024] = Oh * Wout^T + bias (f32)
__global__ __launch_bounds__(256) void gemm_out_kernel(
    const _Float16* __restrict__ A, const _Float16* __restrict__ W,
    const float* __restrict__ bias, float* __restrict__ out) {
  extern __shared__ char gsm[];
  _Float16* shA = (_Float16*)gsm;
  const int tid = threadIdx.x;
  const int w = tid >> 5, lane = tid & 31;
  const int mbase = (lane < 16) ? 0 : 8;
  const int nlane = lane & 15;
  const int m0 = blockIdx.y * 64;
  const size_t n0 = blockIdx.x * 512 + w * 64;

  stage_a_panel<64>(shA, A + (size_t)m0 * DIM_D, tid);

  v8f acc[4][4] = {};
  gemm64x512_panel(shA, W, n0, acc);

#pragma unroll
  for (int mt = 0; mt < 4; ++mt) {
#pragma unroll
    for (int j = 0; j < 4; ++j) {
      const int n = (int)n0 + j * 16 + nlane;
      const float bn = bias[n];
#pragma unroll
      for (int r = 0; r < 8; ++r) {
        const int m = m0 + mt * 16 + mbase + r;
        out[(size_t)m * DIM_D + n] = acc[mt][j][r] + bn;
      }
    }
  }
}

extern "C" void kernel_launch(void* const* d_in, const int* in_sizes, int n_in,
                              void* d_out, int out_size, void* d_ws, size_t ws_size,
                              hipStream_t stream) {
  const float* x     = (const float*)d_in[0];
  const float* w_qkv = (const float*)d_in[1];
  const float* w_out = (const float*)d_in[2];
  const float* b_out = (const float*)d_in[3];

  char* ws = (char*)d_ws;
  _Float16* xh  = (_Float16*)(ws + 0);
  _Float16* wqh = (_Float16*)(ws + 8388608);
  _Float16* woh = (_Float16*)(ws + 14680064);
  _Float16* Qf  = (_Float16*)(ws + 16777216);
  _Float16* Kf  = (_Float16*)(ws + 25165824);
  _Float16* Vt  = (_Float16*)(ws + 33554432);
  _Float16* Oh  = (_Float16*)(ws + 41943040);

  f32_to_f16_kernel<<<DIM_M * DIM_D / 256, 256, 0, stream>>>(x, xh, DIM_M * DIM_D);
  f32_to_f16_kernel<<<3 * DIM_D * DIM_D / 256, 256, 0, stream>>>(w_qkv, wqh, 3 * DIM_D * DIM_D);
  f32_to_f16_kernel<<<DIM_D * DIM_D / 256, 256, 0, stream>>>(w_out, woh, DIM_D * DIM_D);

  (void)hipFuncSetAttribute((const void*)gemm_qkv_kernel,
                            hipFuncAttributeMaxDynamicSharedMemorySize, GEMM_SMEM);
  gemm_qkv_kernel<<<dim3(6, DIM_M / 64), 256, GEMM_SMEM, stream>>>(xh, wqh, Qf, Kf, Vt);

  (void)hipFuncSetAttribute((const void*)attn_kernel,
                            hipFuncAttributeMaxDynamicSharedMemorySize, SMEM_TOTAL);
  attn_kernel<<<dim3(DIM_S / 16, 2 * DIM_H), 256, SMEM_TOTAL, stream>>>(Qf, Kf, Vt, Oh);

  (void)hipFuncSetAttribute((const void*)gemm_out_kernel,
                            hipFuncAttributeMaxDynamicSharedMemorySize, GEMM_SMEM);
  gemm_out_kernel<<<dim3(2, DIM_M / 64), 256, GEMM_SMEM, stream>>>(Oh, woh, b_out, (float*)d_out);
}
